// Net_66408784331557
// MI455X (gfx1250) — compile-verified, matrix operand order checked by reading, not run on hardware
//
#include <hip/hip_runtime.h>

// ---------------- problem constants ----------------
#define B_ROWS   65536
#define NFEAT    2048
#define HID      75
#define NLAYERS  12
#define NT       6              // 6 N-tiles of 16 -> 96 padded columns
#define PITCH    96             // padded activation width (fp32)
#define KC_IN    64             // 2048 / 32 k-chunks for input layer
#define KC_MID   3              // 96 / 32 k-chunks for mid layers
#define NGROUPS  64             // row groups for deterministic column reduction
#define ROWS_PER_GROUP (B_ROWS / NGROUPS)
#define EPSV     1e-5f

typedef __attribute__((ext_vector_type(16))) __bf16 v16bf;
typedef __attribute__((ext_vector_type(8)))  float  v8f;

// Native convert (lowered to v_cvt_pk_bf16_f32 when vectorized); RNE rounding.
__device__ __forceinline__ __bf16 f2bf(float f) { return (__bf16)f; }
__device__ __forceinline__ unsigned short f2bf_bits(float f) {
  union { __bf16 b; unsigned short s; } o; o.b = (__bf16)f;
  return o.s;
}

// ---------------- kernel 1: deterministic column sum / sumsq ----------------
// partial layout: [group][ncols][2]
__global__ void colstats_kernel(const float* __restrict__ A, int pitch, int ncols,
                                float* __restrict__ partial) {
  int c = blockIdx.x * blockDim.x + threadIdx.x;
  if (c >= ncols) return;
  size_t r0 = (size_t)blockIdx.y * ROWS_PER_GROUP;
  float s = 0.f, q = 0.f;
  for (int i = 0; i < ROWS_PER_GROUP; ++i) {
    float v = A[(r0 + i) * (size_t)pitch + c];
    s += v;
    q = fmaf(v, v, q);
  }
  size_t o = ((size_t)blockIdx.y * ncols + c) * 2;
  partial[o]     = s;
  partial[o + 1] = q;
}

// ---------------- kernel 2: finalize BN stats -> scale/shift ----------------
__global__ void finalize_stats_kernel(const float* __restrict__ partial, int ncols, int nvalid,
                                      const float* __restrict__ gamma,
                                      const float* __restrict__ beta,
                                      float* __restrict__ scale, float* __restrict__ shift) {
  int c = blockIdx.x * blockDim.x + threadIdx.x;
  if (c >= ncols) return;
  if (c >= nvalid) { scale[c] = 0.f; shift[c] = 0.f; return; }
  float s = 0.f, q = 0.f;
  for (int g = 0; g < NGROUPS; ++g) {
    size_t o = ((size_t)g * ncols + c) * 2;
    s += partial[o];
    q += partial[o + 1];
  }
  const float invB = 1.0f / (float)B_ROWS;
  float mean = s * invB;
  float var  = fmaxf(q * invB - mean * mean, 0.f);   // biased variance
  float sc   = gamma[c] * rsqrtf(var + EPSV);
  scale[c] = sc;
  shift[c] = beta[c] - mean * sc;
}

// ---------------- kernel 3: fold BN into weights, swizzle to WMMA B layout --
// Wb layout: [kc][nt][lane(32)][e(16)] bf16, matching V_WMMA_F32_16X16X32_BF16
// B-operand VGPR layout (lane -> N, packed K pairs, halves split by lane>=16).
__global__ void fold_weights_kernel(const float* __restrict__ W, int ldw,
                                    int kvalid, int nvalid, int kc_count,
                                    const float* __restrict__ scale,
                                    const float* __restrict__ shift,
                                    const float* __restrict__ bias_in,
                                    unsigned short* __restrict__ Wb,
                                    float* __restrict__ biasf) {
  int total = kc_count * NT * 32 * 16;
  for (int idx = blockIdx.x * blockDim.x + threadIdx.x; idx < total;
       idx += gridDim.x * blockDim.x) {
    int e    = idx & 15;
    int lane = (idx >> 4) & 31;
    int nt   = (idx >> 9) % NT;
    int kc   = (idx >> 9) / NT;
    int hh = lane >> 4, nl = lane & 15;
    int p = e >> 1, bit = e & 1;
    int klocal = ((p & 3) * 2 + bit) + ((p >= 4) ? 16 : 0) + hh * 8;
    int k = kc * 32 + klocal;
    int n = nt * 16 + nl;
    float v = 0.f;
    if (n < nvalid && k < kvalid) v = W[(size_t)n * ldw + k] * scale[k];
    Wb[idx] = f2bf_bits(v);
  }
  // folded bias: b'[j] = b[j] + sum_k W[j,k] * shift[k]   (j in [0, NT*16))
  if (blockIdx.x == 0 && threadIdx.x < NT * 16) {
    int j = threadIdx.x;
    float b = 0.f;
    if (j < nvalid) {
      b = bias_in[j];
      for (int k = 0; k < kvalid; ++k) b = fmaf(W[(size_t)j * ldw + k], shift[k], b);
    }
    biasf[j] = b;
  }
}

// ---------------- kernel 4: WMMA GEMM  out = relu(A @ W'^T + b') -----------
// A: fp32 [B_ROWS][lda]; Wb: swizzled bf16; out: fp32 [B_ROWS][PITCH].
// Block = 256 threads = 8 waves; wave computes 32 rows x 96 cols.
__global__ __launch_bounds__(256)
void mlp_gemm_kernel(const float* __restrict__ A, int lda, int kc_count,
                     const unsigned short* __restrict__ Wb,
                     const float* __restrict__ biasf,
                     float* __restrict__ out, int do_relu) {
  const int lane = threadIdx.x & 31;
  const int wave = threadIdx.x >> 5;
  const int rowBase = blockIdx.x * 256 + wave * 32;
  const int hh = lane >> 4;     // lane half selects K sub-blocks
  const int ml = lane & 15;     // A: row-in-tile, B/C/D: col-in-tile

  v8f acc[2][NT];
#pragma unroll
  for (int t = 0; t < 2; ++t)
#pragma unroll
    for (int nt = 0; nt < NT; ++nt)
#pragma unroll
      for (int r = 0; r < 8; ++r) acc[t][nt][r] = 0.f;

  for (int kc = 0; kc < kc_count; ++kc) {
    v16bf a[2];
#pragma unroll
    for (int t = 0; t < 2; ++t) {
      const float* ap = A + (size_t)(rowBase + t * 16 + ml) * lda + kc * 32 + hh * 8;
      float4 x0 = *(const float4*)(ap);
      float4 x1 = *(const float4*)(ap + 4);
      float4 x2 = *(const float4*)(ap + 16);
      float4 x3 = *(const float4*)(ap + 20);
      float tmp[16] = { x0.x, x0.y, x0.z, x0.w,  x1.x, x1.y, x1.z, x1.w,
                        x2.x, x2.y, x2.z, x2.w,  x3.x, x3.y, x3.z, x3.w };
#pragma unroll
      for (int e = 0; e < 16; ++e) a[t][e] = f2bf(tmp[e]);
    }
#pragma unroll
    for (int nt = 0; nt < NT; ++nt) {
      v16bf b = *(const v16bf*)(Wb + (((size_t)kc * NT + nt) * 32 + lane) * 16);
      acc[0][nt] = __builtin_amdgcn_wmma_f32_16x16x32_bf16(
          false, a[0], false, b, (short)0, acc[0][nt], false, false);
      acc[1][nt] = __builtin_amdgcn_wmma_f32_16x16x32_bf16(
          false, a[1], false, b, (short)0, acc[1][nt], false, false);
    }
  }

  // C/D layout: VGPR r -> row (r + 8*hh), lane%16 -> col
#pragma unroll
  for (int nt = 0; nt < NT; ++nt) {
    float bv = biasf[nt * 16 + ml];
#pragma unroll
    for (int t = 0; t < 2; ++t) {
#pragma unroll
      for (int r = 0; r < 8; ++r) {
        float v = acc[t][nt][r] + bv;
        if (do_relu) v = fmaxf(v, 0.f);
        out[(size_t)(rowBase + t * 16 + hh * 8 + r) * PITCH + nt * 16 + ml] = v;
      }
    }
  }
}

// ---------------- kernel 5: output head (75 -> 1) ---------------------------
__global__ void head_kernel(const float* __restrict__ H,
                            const float* __restrict__ Wout,
                            const float* __restrict__ bout,
                            float* __restrict__ out) {
  int r = blockIdx.x * blockDim.x + threadIdx.x;
  if (r >= B_ROWS) return;
  float s = bout[0];
  for (int k = 0; k < HID; ++k) s = fmaf(H[(size_t)r * PITCH + k], Wout[k], s);
  out[r] = s;
}

// ---------------- host launch ----------------------------------------------
extern "C" void kernel_launch(void* const* d_in, const int* in_sizes, int n_in,
                              void* d_out, int out_size, void* d_ws, size_t ws_size,
                              hipStream_t stream) {
  const float* x        = (const float*)d_in[0];
  const float* bn_g_in  = (const float*)d_in[1];
  const float* bn_b_in  = (const float*)d_in[2];
  const float* W_in     = (const float*)d_in[3];   // [75][2048]
  const float* b_in     = (const float*)d_in[4];
  const float* mid_g    = (const float*)d_in[5];   // [12][75]
  const float* mid_b    = (const float*)d_in[6];
  const float* mid_W    = (const float*)d_in[7];   // [12][75][75]
  const float* mid_bias = (const float*)d_in[8];
  const float* W_out    = (const float*)d_in[9];   // [1][75]
  const float* b_out    = (const float*)d_in[10];
  float* out = (float*)d_out;

  // workspace carve-up (256B aligned)
  char* ws = (char*)d_ws;
  size_t off = 0;
  auto take = [&](size_t bytes) { void* p = ws + off; off = (off + bytes + 255) & ~(size_t)255; return p; };
  float* partial  = (float*)take((size_t)NGROUPS * NFEAT * 2 * sizeof(float)); // reused for h stats
  float* scale    = (float*)take(NFEAT * sizeof(float));
  float* shift    = (float*)take(NFEAT * sizeof(float));
  float* biasf    = (float*)take(NT * 16 * sizeof(float));
  unsigned short* Wb_in  = (unsigned short*)take((size_t)KC_IN  * NT * 512 * sizeof(unsigned short));
  unsigned short* Wb_mid = (unsigned short*)take((size_t)KC_MID * NT * 512 * sizeof(unsigned short));
  float* h0 = (float*)take((size_t)B_ROWS * PITCH * sizeof(float));
  float* h1 = (float*)take((size_t)B_ROWS * PITCH * sizeof(float));
  (void)ws_size; (void)n_in; (void)in_sizes; (void)out_size;

  // ---- input block: stats(x) -> fold BN into W_in -> WMMA GEMM + ReLU ----
  colstats_kernel<<<dim3((NFEAT + 255) / 256, NGROUPS), 256, 0, stream>>>(
      x, NFEAT, NFEAT, partial);
  finalize_stats_kernel<<<(NFEAT + 255) / 256, 256, 0, stream>>>(
      partial, NFEAT, NFEAT, bn_g_in, bn_b_in, scale, shift);
  fold_weights_kernel<<<(KC_IN * NT * 512 + 255) / 256, 256, 0, stream>>>(
      W_in, NFEAT, NFEAT, HID, KC_IN, scale, shift, b_in, Wb_in, biasf);
  mlp_gemm_kernel<<<B_ROWS / 256, 256, 0, stream>>>(
      x, NFEAT, KC_IN, Wb_in, biasf, h0, 1);

  // ---- 12 middle blocks (all L2-resident) ----
  float* hc = h0;
  float* hn = h1;
  for (int L = 0; L < NLAYERS; ++L) {
    colstats_kernel<<<dim3(1, NGROUPS), 256, 0, stream>>>(hc, PITCH, PITCH, partial);
    finalize_stats_kernel<<<1, 256, 0, stream>>>(
        partial, PITCH, HID, mid_g + L * HID, mid_b + L * HID, scale, shift);
    fold_weights_kernel<<<(KC_MID * NT * 512 + 255) / 256, 256, 0, stream>>>(
        mid_W + (size_t)L * HID * HID, HID, HID, HID, KC_MID,
        scale, shift, mid_bias + L * HID, Wb_mid, biasf);
    mlp_gemm_kernel<<<B_ROWS / 256, 256, 0, stream>>>(
        hc, PITCH, KC_MID, Wb_mid, biasf, hn, 1);
    float* t = hc; hc = hn; hn = t;
  }

  // ---- output head ----
  head_kernel<<<B_ROWS / 256, 256, 0, stream>>>(hc, W_out, b_out, out);
}
